// Informer_45079976739603
// MI455X (gfx1250) — compile-verified, matrix operand order checked by reading, not run on hardware
//
#include <hip/hip_runtime.h>
#include <math.h>

// ============================================================================
// Informer forward for MI455X (gfx1250, wave32, WMMA).
// All dense matmuls -> v_wmma_f32_16x16x32_f16 (f32 accumulate).
// GEMM: register double-buffered LDS staging (global b128 loads overlap WMMA),
//       4x4 register transpose for the B tile -> packed ds_store_b64.
// Attention -> fused flash-style WMMA kernel (scores + softmax + P@V),
// with block-shared V tile staged in LDS (f16) and pipelined K loads.
// ============================================================================

#define D_MODEL 512
#define N_HEADS 8
#define DH      64
#define BATCH   16

typedef __attribute__((ext_vector_type(16))) _Float16 v16h;
typedef __attribute__((ext_vector_type(4)))  _Float16 v4h;
typedef __attribute__((ext_vector_type(8)))  float    v8f;

// ---------------------------------------------------------------------------
// Generic tiled WMMA GEMM:  C[M,N] = act(A[M,K] @ W[K,N] + bias[N])
// Block = 256 threads (8 waves). Block tile 128x128, wave tile 32x64,
// K stepped by 32 through LDS (f32 -> f16 convert on store to LDS).
// act: 0 = none, 1 = exact GELU (erf), 2 = ReLU.  Requires K % 32 == 0.
// ---------------------------------------------------------------------------
#define TM 128
#define TN 128
#define TK 32
#define KPAD 4

__device__ __forceinline__ float f4c(const float4& v, int j) {
    return j == 0 ? v.x : j == 1 ? v.y : j == 2 ? v.z : v.w;
}

__global__ __launch_bounds__(256) void wmma_gemm_kernel(
    const float* __restrict__ A, const float* __restrict__ Bw,
    const float* __restrict__ bias, float* __restrict__ C,
    int M, int N, int K, int act)
{
    __shared__ _Float16 As[TM][TK + KPAD];
    __shared__ _Float16 Bs[TN][TK + KPAD];

    const int tid  = threadIdx.x;
    const int lane = tid & 31;
    const int wave = tid >> 5;
    const int half = lane >> 4;   // 0: lanes 0-15, 1: lanes 16-31
    const int l16  = lane & 15;
    const int kb   = half * 8;    // K sub-offset per ISA A/B layout

    const int bm = blockIdx.y * TM;
    const int bn = blockIdx.x * TN;
    const int wm = (wave >> 1) * 32;   // 4 wave-rows of 32
    const int wn = (wave & 1) * 64;    // 2 wave-cols of 64

    // Uniform fast path: full tile, vector-safe N (true for all but proj2)
    const bool full = ((N & 3) == 0) && (bm + TM <= M) && (bn + TN <= N);

    // Per-thread staging coordinates (4 float4 each for A and B)
    const int am  = tid >> 3;          // A row   = am + i*32
    const int ak4 = (tid & 7) << 2;    // A k-off
    const int bkg = (tid >> 5) << 2;   // B k rows = bkg + 0..3 (consecutive)
    const int bn4 = (tid & 31) << 2;   // B n-off

    float4 ra[4], rb[4];

    auto load_tiles = [&](int k0) {
        if (full) {
#pragma unroll
            for (int i = 0; i < 4; ++i)
                ra[i] = *(const float4*)(A + (size_t)(bm + am + i * 32) * K + k0 + ak4);
#pragma unroll
            for (int i = 0; i < 4; ++i)
                rb[i] = *(const float4*)(Bw + (size_t)(k0 + bkg + i) * N + bn + bn4);
        } else {
#pragma unroll
            for (int i = 0; i < 4; ++i) {
                int gm = bm + am + i * 32;
                float4 v = {0.f, 0.f, 0.f, 0.f};
                if (gm < M) v = *(const float4*)(A + (size_t)gm * K + k0 + ak4);
                ra[i] = v;
            }
#pragma unroll
            for (int i = 0; i < 4; ++i) {
                const float* bp = Bw + (size_t)(k0 + bkg + i) * N;
                int gn = bn + bn4;
                float4 v = {0.f, 0.f, 0.f, 0.f};
                if (gn     < N) v.x = bp[gn];
                if (gn + 1 < N) v.y = bp[gn + 1];
                if (gn + 2 < N) v.z = bp[gn + 2];
                if (gn + 3 < N) v.w = bp[gn + 3];
                rb[i] = v;
            }
        }
    };

    auto store_tiles = [&]() {
#pragma unroll
        for (int i = 0; i < 4; ++i) {
            v4h hv = { (_Float16)ra[i].x, (_Float16)ra[i].y,
                       (_Float16)ra[i].z, (_Float16)ra[i].w };
            *(v4h*)&As[am + i * 32][ak4] = hv;
        }
        // 4x4 register transpose: rows = consecutive k, cols = n -> Bs[n][k]
#pragma unroll
        for (int j = 0; j < 4; ++j) {
            v4h hv = { (_Float16)f4c(rb[0], j), (_Float16)f4c(rb[1], j),
                       (_Float16)f4c(rb[2], j), (_Float16)f4c(rb[3], j) };
            *(v4h*)&Bs[bn4 + j][bkg] = hv;
        }
    };

    v8f acc[2][4] = {};

    load_tiles(0);
    for (int k0 = 0; k0 < K; k0 += TK) {
        store_tiles();
        __syncthreads();
        // Issue next tile's global loads; consumed only at next store_tiles(),
        // so their latency hides behind the fragment reads + WMMAs below.
        if (k0 + TK < K) load_tiles(k0 + TK);

        v16h af[2], bf[4];
#pragma unroll
        for (int i = 0; i < 2; ++i) {
            const _Float16* r = &As[wm + i * 16 + l16][0];
#pragma unroll
            for (int e = 0; e < 8; ++e) { af[i][e] = r[kb + e]; af[i][8 + e] = r[16 + kb + e]; }
        }
#pragma unroll
        for (int j = 0; j < 4; ++j) {
            const _Float16* r = &Bs[wn + j * 16 + l16][0];
#pragma unroll
            for (int e = 0; e < 8; ++e) { bf[j][e] = r[kb + e]; bf[j][8 + e] = r[16 + kb + e]; }
        }
#pragma unroll
        for (int i = 0; i < 2; ++i)
#pragma unroll
            for (int j = 0; j < 4; ++j)
                acc[i][j] = __builtin_amdgcn_wmma_f32_16x16x32_f16(
                    false, af[i], false, bf[j], (short)0, acc[i][j], false, false);
        __syncthreads();
    }

    // Epilogue: C/D layout -> M = r + 8*half, N = l16 within each 16x16 tile
#pragma unroll
    for (int i = 0; i < 2; ++i) {
#pragma unroll
        for (int j = 0; j < 4; ++j) {
#pragma unroll
            for (int r = 0; r < 8; ++r) {
                int gm = bm + wm + i * 16 + half * 8 + r;
                int gn = bn + wn + j * 16 + l16;
                if (gm < M && gn < N) {
                    float v = acc[i][j][r];
                    if (bias) v += bias[gn];
                    if (act == 1)      v = 0.5f * v * (1.0f + erff(v * 0.70710678118f));
                    else if (act == 2) v = fmaxf(v, 0.0f);
                    C[(size_t)gm * N + gn] = v;
                }
            }
        }
    }
}

// ---------------------------------------------------------------------------
// Fused attention (flash-style, full attention; causal flag for decoder
// self-attn, 'mix' selects Informer's transposed write-out).
// Q,K,V in [B, L, H, 64] (i.e. [B*L, 512]).  One wave per 16-row Q tile.
// Both waves share (b,h): V tile staged once per block into LDS (f16).
// ---------------------------------------------------------------------------
#define AW 2  // waves per block

__global__ __launch_bounds__(64) void attention_kernel(
    const float* __restrict__ Q, const float* __restrict__ K,
    const float* __restrict__ V, float* __restrict__ O,
    int Lq, int Lk, int causal, int mix)
{
    __shared__ float    sc[AW][16][512];   // per-wave score rows (64 KB)
    __shared__ _Float16 Vs[512][DH + 2];   // shared V tile, f16 (~67 KB)

    const int tid  = threadIdx.x;
    const int lane = tid & 31;
    const int wave = tid >> 5;
    const int half = lane >> 4;
    const int l16  = lane & 15;
    const int kb   = half * 8;

    const int b  = blockIdx.x >> 3;
    const int h  = blockIdx.x & 7;

    // ---- block-cooperative V staging (coalesced float4, f32 -> f16) ----
    for (int f = tid; f < Lk * (DH / 4); f += 32 * AW) {
        int s  = f >> 4;             // 16 float4 per 64-wide row
        int d4 = (f & 15) << 2;
        float4 v = *(const float4*)(V + ((size_t)(b * Lk + s) * N_HEADS + h) * DH + d4);
        v4h hv = { (_Float16)v.x, (_Float16)v.y, (_Float16)v.z, (_Float16)v.w };
        *(v4h*)&Vs[s][d4] = hv;
    }
    __syncthreads();

    const int qt = blockIdx.y * AW + wave;   // uniform per wave
    if (qt * 16 >= Lq) return;

    auto cvt16 = [&](const float4* r, v16h& dst) {
        dst[0]  = (_Float16)r[0].x; dst[1]  = (_Float16)r[0].y;
        dst[2]  = (_Float16)r[0].z; dst[3]  = (_Float16)r[0].w;
        dst[4]  = (_Float16)r[1].x; dst[5]  = (_Float16)r[1].y;
        dst[6]  = (_Float16)r[1].z; dst[7]  = (_Float16)r[1].w;
        dst[8]  = (_Float16)r[2].x; dst[9]  = (_Float16)r[2].y;
        dst[10] = (_Float16)r[2].z; dst[11] = (_Float16)r[2].w;
        dst[12] = (_Float16)r[3].x; dst[13] = (_Float16)r[3].y;
        dst[14] = (_Float16)r[3].z; dst[15] = (_Float16)r[3].w;
    };

    // ---- Q fragments (A matrix, 16x64 over two K=32 chunks) ----
    v16h qf[2];
    {
        const float* qp = Q + ((size_t)(b * Lq + qt * 16 + l16) * N_HEADS + h) * DH;
#pragma unroll
        for (int i = 0; i < 2; ++i) {
            float4 r[4];
            r[0] = *(const float4*)(qp + i * 32 + kb);
            r[1] = *(const float4*)(qp + i * 32 + kb + 4);
            r[2] = *(const float4*)(qp + i * 32 + 16 + kb);
            r[3] = *(const float4*)(qp + i * 32 + 16 + kb + 4);
            cvt16(r, qf[i]);
        }
    }

    const float scale = 0.125f;  // 1/sqrt(64)
    const int   nst   = Lk / 16;

    auto ldK = [&](int st, float4* r) {
        const float* kp = K + ((size_t)(b * Lk + st * 16 + l16) * N_HEADS + h) * DH;
#pragma unroll
        for (int i = 0; i < 2; ++i) {
            r[i * 4 + 0] = *(const float4*)(kp + i * 32 + kb);
            r[i * 4 + 1] = *(const float4*)(kp + i * 32 + kb + 4);
            r[i * 4 + 2] = *(const float4*)(kp + i * 32 + 16 + kb);
            r[i * 4 + 3] = *(const float4*)(kp + i * 32 + 16 + kb + 4);
        }
    };

    // ---- scores = Q @ K^T (pipelined K-tile loads) ----
    float4 kr[8];
    ldK(0, kr);
    for (int st = 0; st < nst; ++st) {
        v16h kf[2];
        cvt16(&kr[0], kf[0]);
        cvt16(&kr[4], kf[1]);
        if (st + 1 < nst) ldK(st + 1, kr);   // overlap with WMMA + stores

        v8f s = {};
        s = __builtin_amdgcn_wmma_f32_16x16x32_f16(false, qf[0], false, kf[0], (short)0, s, false, false);
        s = __builtin_amdgcn_wmma_f32_16x16x32_f16(false, qf[1], false, kf[1], (short)0, s, false, false);
#pragma unroll
        for (int r = 0; r < 8; ++r) {
            int qrow = qt * 16 + half * 8 + r;
            int kcol = st * 16 + l16;
            float v = s[r] * scale;
            if (causal && kcol > qrow) v = -1e30f;
            sc[wave][half * 8 + r][kcol] = v;
        }
    }

    // ---- softmax over Lk (lane pair per row, combine via shuffle) ----
    {
        const int c0 = half * (Lk / 2), c1 = c0 + Lk / 2;
        float mx = -1e30f;
        for (int c = c0; c < c1; ++c) mx = fmaxf(mx, sc[wave][l16][c]);
        mx = fmaxf(mx, __shfl_xor(mx, 16));
        float sum = 0.0f;
        for (int c = c0; c < c1; ++c) {
            float p = __expf(sc[wave][l16][c] - mx);
            sc[wave][l16][c] = p;
            sum += p;
        }
        sum += __shfl_xor(sum, 16);
        float inv = 1.0f / sum;
        for (int c = c0; c < c1; ++c) sc[wave][l16][c] *= inv;
    }

    // ---- context = P @ V (V from LDS) ----
    v8f oacc[4] = {};
    for (int s0 = 0; s0 < Lk; s0 += 32) {
        v16h pf;
        {
            const float* pr = &sc[wave][l16][s0];
            float4 r[4];
            r[0] = *(const float4*)(pr + kb);
            r[1] = *(const float4*)(pr + kb + 4);
            r[2] = *(const float4*)(pr + 16 + kb);
            r[3] = *(const float4*)(pr + 16 + kb + 4);
            cvt16(r, pf);
        }
#pragma unroll
        for (int j = 0; j < 4; ++j) {
            v16h vf;
            const int d = j * 16 + l16;
#pragma unroll
            for (int e = 0; e < 8; ++e) {
                vf[e]     = Vs[s0 + kb + e][d];
                vf[8 + e] = Vs[s0 + 16 + kb + e][d];
            }
            oacc[j] = __builtin_amdgcn_wmma_f32_16x16x32_f16(false, pf, false, vf, (short)0, oacc[j], false, false);
        }
    }

    // ---- write out (mix = Informer transposed layout) ----
#pragma unroll
    for (int j = 0; j < 4; ++j)
#pragma unroll
        for (int r = 0; r < 8; ++r) {
            int qrow = qt * 16 + half * 8 + r;
            int d    = j * 16 + l16;
            size_t off;
            if (mix) off = ((size_t)b * Lq * N_HEADS + (size_t)h * Lq + qrow) * DH + d;
            else     off = ((size_t)(b * Lq + qrow) * N_HEADS + h) * DH + d;
            O[off] = oacc[j][r];
        }
}

// ---------------------------------------------------------------------------
// LayerNorm with optional residual: Y = g * norm(X + R) + b.  One wave/row.
// ---------------------------------------------------------------------------
__global__ __launch_bounds__(256) void layernorm_kernel(
    const float* __restrict__ X, const float* __restrict__ R,
    const float* __restrict__ g, const float* __restrict__ bta,
    float* __restrict__ Y, int rows)
{
    int lane = threadIdx.x & 31;
    int wave = threadIdx.x >> 5;
    int row  = blockIdx.x * 8 + wave;
    if (row >= rows) return;
    const float* xr = X + (size_t)row * D_MODEL;
    const float* rr = R ? R + (size_t)row * D_MODEL : nullptr;
    float v[16];
    float s = 0.f;
#pragma unroll
    for (int i = 0; i < 16; ++i) {
        int c = lane + i * 32;
        float t = xr[c] + (rr ? rr[c] : 0.f);
        v[i] = t; s += t;
    }
#pragma unroll
    for (int o = 16; o > 0; o >>= 1) s += __shfl_xor(s, o);
    float mean = s * (1.0f / D_MODEL);
    float vs = 0.f;
#pragma unroll
    for (int i = 0; i < 16; ++i) { float d = v[i] - mean; vs += d * d; }
#pragma unroll
    for (int o = 16; o > 0; o >>= 1) vs += __shfl_xor(vs, o);
    float rinv = rsqrtf(vs * (1.0f / D_MODEL) + 1e-5f);
    float* yr = Y + (size_t)row * D_MODEL;
#pragma unroll
    for (int i = 0; i < 16; ++i) {
        int c = lane + i * 32;
        yr[c] = g[c] * ((v[i] - mean) * rinv) + bta[c];
    }
}

// ---------------------------------------------------------------------------
// Data embedding: circular conv1d(k=3, Cin=7) + sinusoidal PE + time linear.
// ---------------------------------------------------------------------------
__global__ void embed_kernel(
    const float* __restrict__ x, const float* __restrict__ xm,
    const float* __restrict__ cw, const float* __restrict__ cb,
    const float* __restrict__ tw, const float* __restrict__ tb,
    float* __restrict__ y, int L, int total)
{
    int idx = blockIdx.x * blockDim.x + threadIdx.x;
    if (idx >= total) return;
    int d = idx & (D_MODEL - 1);
    int l = (idx >> 9) % L;
    int b = idx / (D_MODEL * L);
    float acc = cb[d] + tb[d];
#pragma unroll
    for (int t = 0; t < 3; ++t) {
        int ls = l + t - 1;
        ls = (ls < 0) ? (L - 1) : ((ls >= L) ? 0 : ls);
        const float* xp = x + (size_t)(b * L + ls) * 7;
        const float* wp = cw + (size_t)t * 7 * D_MODEL + d;
#pragma unroll
        for (int c = 0; c < 7; ++c) acc += xp[c] * wp[c * D_MODEL];
    }
    {   // positional embedding
        int pair = d >> 1;
        float freq = __expf(-(float)(2 * pair) * (9.210340372f / (float)D_MODEL));
        float ang = (float)l * freq;
        acc += (d & 1) ? cosf(ang) : sinf(ang);
    }
    const float* mp = xm + (size_t)(b * L + l) * 4;
#pragma unroll
    for (int j = 0; j < 4; ++j) acc += mp[j] * tw[j * D_MODEL + d];
    y[idx] = acc;
}

// ---------------------------------------------------------------------------
// im2col for distilling conv (k=3 circular, 512ch), float4-vectorized.
// ---------------------------------------------------------------------------
__global__ void im2col_kernel(const float* __restrict__ x, float* __restrict__ y,
                              int L, int total4)
{
    int idx = blockIdx.x * blockDim.x + threadIdx.x;
    if (idx >= total4) return;
    int col4 = idx % 384;           // 1536/4
    int rowi = idx / 384;
    int l = rowi % L, b = rowi / L;
    int t = col4 >> 7;              // /128
    int d = (col4 & 127) << 2;
    int ls = l + t - 1;
    ls = (ls < 0) ? (L - 1) : ((ls >= L) ? 0 : ls);
    float4 v = *(const float4*)(x + (size_t)(b * L + ls) * D_MODEL + d);
    *(float4*)(y + (size_t)idx * 4) = v;
}

// ---------------------------------------------------------------------------
// scale + ELU + pad(-1e30) + maxpool(k=3, s=2): [B,L,512] -> [B,L/2,512]
// ---------------------------------------------------------------------------
__global__ void elu_pool_kernel(const float* __restrict__ x, float* __restrict__ y,
                                int L, int total4)
{
    int idx = blockIdx.x * blockDim.x + threadIdx.x;
    if (idx >= total4) return;
    int Lo = L >> 1;
    int d = (idx & 127) << 2;
    int i = (idx >> 7) % Lo;
    int b = idx / (128 * Lo);
    const float inv = 0.999995000037f;  // 1/sqrt(1+1e-5)
    float4 m = {-1e30f, -1e30f, -1e30f, -1e30f};
#pragma unroll
    for (int t = -1; t <= 1; ++t) {
        int l = 2 * i + t;
        if (l >= 0 && l < L) {
            float4 v = *(const float4*)(x + (size_t)(b * L + l) * D_MODEL + d);
            v.x *= inv; v.y *= inv; v.z *= inv; v.w *= inv;
            v.x = (v.x > 0.f) ? v.x : (__expf(v.x) - 1.0f);
            v.y = (v.y > 0.f) ? v.y : (__expf(v.y) - 1.0f);
            v.z = (v.z > 0.f) ? v.z : (__expf(v.z) - 1.0f);
            v.w = (v.w > 0.f) ? v.w : (__expf(v.w) - 1.0f);
            m.x = fmaxf(m.x, v.x); m.y = fmaxf(m.y, v.y);
            m.z = fmaxf(m.z, v.z); m.w = fmaxf(m.w, v.w);
        }
    }
    *(float4*)(y + (size_t)idx * 4) = m;
}

// ---------------------------------------------------------------------------
// Final slice: out[b,p,c] = proj2[b, 256+p, c]
// ---------------------------------------------------------------------------
__global__ void slice_kernel(const float* __restrict__ x, float* __restrict__ y,
                             int total)
{
    int idx = blockIdx.x * blockDim.x + threadIdx.x;
    if (idx >= total) return;
    int c = idx % 7;
    int p = (idx / 7) % 256;
    int b = idx / (7 * 256);
    y[idx] = x[(size_t)(b * 512 + 256 + p) * 7 + c];
}

// ===========================================================================
// Host orchestration
// ===========================================================================
struct AttnW { const float *wq, *bq, *wk, *bk, *wv, *bv, *wo, *bo; };

extern "C" void kernel_launch(void* const* d_in, const int* in_sizes, int n_in,
                              void* d_out, int out_size, void* d_ws, size_t ws_size,
                              hipStream_t stream)
{
    (void)in_sizes; (void)n_in; (void)out_size; (void)ws_size;
    auto in = [&](int i) { return (const float*)d_in[i]; };
    int ix = 0;
    const float* x_enc      = in(ix++);
    const float* x_mark_enc = in(ix++);
    const float* x_dec      = in(ix++);
    const float* x_mark_dec = in(ix++);
    // params, recursive insertion order:
    // enc_emb, dec_emb, enc_layers, conv_layers, enc_norm, dec_layers,
    // dec_norm, proj1, proj2
    const float *e_cw = in(ix++), *e_cb = in(ix++), *e_tw = in(ix++), *e_tb = in(ix++);
    const float *d_cw = in(ix++), *d_cb = in(ix++), *d_tw = in(ix++), *d_tb = in(ix++);

    struct EncL { AttnW a; const float *c1w,*c1b,*c2w,*c2b,*n1g,*n1b,*n2g,*n2b; } enc[3];
    for (int i = 0; i < 3; ++i) {
        enc[i].a.wq = in(ix++); enc[i].a.bq = in(ix++);
        enc[i].a.wk = in(ix++); enc[i].a.bk = in(ix++);
        enc[i].a.wv = in(ix++); enc[i].a.bv = in(ix++);
        enc[i].a.wo = in(ix++); enc[i].a.bo = in(ix++);
        enc[i].c1w = in(ix++); enc[i].c1b = in(ix++);
        enc[i].c2w = in(ix++); enc[i].c2b = in(ix++);
        enc[i].n1g = in(ix++); enc[i].n1b = in(ix++);
        enc[i].n2g = in(ix++); enc[i].n2b = in(ix++);
    }
    const float *cl_w[2], *cl_b[2];
    for (int i = 0; i < 2; ++i) { cl_w[i] = in(ix++); cl_b[i] = in(ix++); }
    const float *en_g = in(ix++), *en_b = in(ix++);

    struct DecL { AttnW s, c; const float *c1w,*c1b,*c2w,*c2b,*n1g,*n1b,*n2g,*n2b,*n3g,*n3b; } dec[2];
    for (int i = 0; i < 2; ++i) {
        dec[i].s.wq = in(ix++); dec[i].s.bq = in(ix++);
        dec[i].s.wk = in(ix++); dec[i].s.bk = in(ix++);
        dec[i].s.wv = in(ix++); dec[i].s.bv = in(ix++);
        dec[i].s.wo = in(ix++); dec[i].s.bo = in(ix++);
        dec[i].c.wq = in(ix++); dec[i].c.bq = in(ix++);
        dec[i].c.wk = in(ix++); dec[i].c.bk = in(ix++);
        dec[i].c.wv = in(ix++); dec[i].c.bv = in(ix++);
        dec[i].c.wo = in(ix++); dec[i].c.bo = in(ix++);
        dec[i].c1w = in(ix++); dec[i].c1b = in(ix++);
        dec[i].c2w = in(ix++); dec[i].c2b = in(ix++);
        dec[i].n1g = in(ix++); dec[i].n1b = in(ix++);
        dec[i].n2g = in(ix++); dec[i].n2b = in(ix++);
        dec[i].n3g = in(ix++); dec[i].n3b = in(ix++);
    }
    const float *dn_g = in(ix++), *dn_b = in(ix++);
    const float *p1w = in(ix++), *p1b = in(ix++);
    const float *p2w = in(ix++), *p2b = in(ix++);

    // ---- workspace bump allocator ----
    const size_t RMAX = (size_t)BATCH * 512;  // 8192 rows max
    char* cur = (char*)d_ws;
    auto alloc = [&](size_t elems) {
        float* p = (float*)cur;
        cur += ((elems * sizeof(float) + 255) & ~(size_t)255);
        return p;
    };
    float* X0 = alloc(RMAX * D_MODEL);
    float* X1 = alloc(RMAX * D_MODEL);
    float* Qb = alloc(RMAX * D_MODEL);
    float* Kb = alloc(RMAX * D_MODEL);
    float* Vb = alloc(RMAX * D_MODEL);
    float* T1 = alloc(RMAX * D_MODEL);
    float* T2 = alloc(RMAX * D_MODEL);
    float* IM = alloc(RMAX * 1536);
    float* EOUT = alloc((size_t)BATCH * 128 * D_MODEL);

    auto gemm = [&](const float* A, const float* W, const float* bias, float* C,
                    int M, int N, int K, int act) {
        dim3 grid((N + TN - 1) / TN, (M + TM - 1) / TM);
        wmma_gemm_kernel<<<grid, 256, 0, stream>>>(A, W, bias, C, M, N, K, act);
    };
    auto attn = [&](const float* Qp, const float* Kp, const float* Vp, float* Op,
                    int Lq, int Lk, int causal, int mix) {
        dim3 grid(BATCH * N_HEADS, (Lq / 16 + AW - 1) / AW);
        attention_kernel<<<grid, 32 * AW, 0, stream>>>(Qp, Kp, Vp, Op, Lq, Lk, causal, mix);
    };
    auto ln = [&](const float* X, const float* R, const float* g, const float* b,
                  float* Y, int rows) {
        layernorm_kernel<<<(rows + 7) / 8, 256, 0, stream>>>(X, R, g, b, Y, rows);
    };

    // ================= Encoder =================
    int L = 512;
    {
        int total = BATCH * L * D_MODEL;
        embed_kernel<<<(total + 255) / 256, 256, 0, stream>>>(
            x_enc, x_mark_enc, e_cw, e_cb, e_tw, e_tb, X0, L, total);
    }
    float* X = X0; float* Y = X1;
    for (int i = 0; i < 3; ++i) {
        int rows = BATCH * L;
        gemm(X, enc[i].a.wq, enc[i].a.bq, Qb, rows, D_MODEL, D_MODEL, 0);
        gemm(X, enc[i].a.wk, enc[i].a.bk, Kb, rows, D_MODEL, D_MODEL, 0);
        gemm(X, enc[i].a.wv, enc[i].a.bv, Vb, rows, D_MODEL, D_MODEL, 0);
        attn(Qb, Kb, Vb, T1, L, L, /*causal=*/0, /*mix=*/0);
        gemm(T1, enc[i].a.wo, enc[i].a.bo, T2, rows, D_MODEL, D_MODEL, 0);
        ln(X, T2, enc[i].n1g, enc[i].n1b, Y, rows);              // Y = LN(X + attn)
        gemm(Y, enc[i].c1w, enc[i].c1b, T1, rows, D_MODEL, D_MODEL, 1);  // GELU
        gemm(T1, enc[i].c2w, enc[i].c2b, T2, rows, D_MODEL, D_MODEL, 0);
        ln(Y, T2, enc[i].n2g, enc[i].n2b, X, rows);              // X = LN(Y + ffn)
        if (i < 2) {
            int totalc4 = rows * 1536 / 4;
            im2col_kernel<<<(totalc4 + 255) / 256, 256, 0, stream>>>(X, IM, L, totalc4);
            gemm(IM, cl_w[i], cl_b[i], T1, rows, D_MODEL, 1536, 0);
            int totalp4 = BATCH * (L / 2) * D_MODEL / 4;
            elu_pool_kernel<<<(totalp4 + 255) / 256, 256, 0, stream>>>(T1, Y, L, totalp4);
            float* t = X; X = Y; Y = t;
            L >>= 1;
        }
    }
    ln(X, nullptr, en_g, en_b, EOUT, BATCH * L);   // L == 128
    const int Le = L, rows_e = BATCH * Le;

    // ================= Decoder =================
    const int Ld = 512, rows_d = BATCH * Ld;
    {
        int total = rows_d * D_MODEL;
        embed_kernel<<<(total + 255) / 256, 256, 0, stream>>>(
            x_dec, x_mark_dec, d_cw, d_cb, d_tw, d_tb, X0, Ld, total);
    }
    float* Dc = X0; float* Dn = X1;
    for (int i = 0; i < 2; ++i) {
        // masked self-attention (prob-causal approximated by full causal), mix=true
        gemm(Dc, dec[i].s.wq, dec[i].s.bq, Qb, rows_d, D_MODEL, D_MODEL, 0);
        gemm(Dc, dec[i].s.wk, dec[i].s.bk, Kb, rows_d, D_MODEL, D_MODEL, 0);
        gemm(Dc, dec[i].s.wv, dec[i].s.bv, Vb, rows_d, D_MODEL, D_MODEL, 0);
        attn(Qb, Kb, Vb, T1, Ld, Ld, /*causal=*/1, /*mix=*/1);
        gemm(T1, dec[i].s.wo, dec[i].s.bo, T2, rows_d, D_MODEL, D_MODEL, 0);
        ln(Dc, T2, dec[i].n1g, dec[i].n1b, Dn, rows_d);
        // cross-attention with encoder output
        gemm(Dn,   dec[i].c.wq, dec[i].c.bq, Qb, rows_d, D_MODEL, D_MODEL, 0);
        gemm(EOUT, dec[i].c.wk, dec[i].c.bk, Kb, rows_e, D_MODEL, D_MODEL, 0);
        gemm(EOUT, dec[i].c.wv, dec[i].c.bv, Vb, rows_e, D_MODEL, D_MODEL, 0);
        attn(Qb, Kb, Vb, T1, Ld, Le, /*causal=*/0, /*mix=*/0);
        gemm(T1, dec[i].c.wo, dec[i].c.bo, T2, rows_d, D_MODEL, D_MODEL, 0);
        ln(Dn, T2, dec[i].n2g, dec[i].n2b, Dc, rows_d);
        // FFN
        gemm(Dc, dec[i].c1w, dec[i].c1b, T1, rows_d, D_MODEL, D_MODEL, 1);   // GELU
        gemm(T1, dec[i].c2w, dec[i].c2b, T2, rows_d, D_MODEL, D_MODEL, 0);
        ln(Dc, T2, dec[i].n3g, dec[i].n3b, Dn, rows_d);
        float* t = Dc; Dc = Dn; Dn = t;
    }
    ln(Dc, nullptr, dn_g, dn_b, Dn, rows_d);
    gemm(Dn, p1w, p1b, T2, rows_d, 256, D_MODEL, 2);   // ReLU, [8192,256]
    gemm(T2, p2w, p2b, T1, rows_d, 7, 256, 0);         // [8192,7]
    {
        int total = BATCH * 256 * 7;
        slice_kernel<<<(total + 255) / 256, 256, 0, stream>>>(T1, (float*)d_out, total);
    }
}
